// ASSA_76630806495892
// MI455X (gfx1250) — compile-verified, matrix operand order checked by reading, not run on hardware
//
#include <hip/hip_runtime.h>
#include <hip/hip_bf16.h>

// Problem constants (from reference): B=4, C=256, H=W=64 -> N=4096 tokens.
#define BB 4
#define CC 256
#define NN 4096
#define BN (BB * NN)
#define EPS 1e-5f

typedef __attribute__((ext_vector_type(16))) __bf16 v16bf;
typedef __attribute__((ext_vector_type(8)))  float  v8f;

// ---------------------------------------------------------------------------
// 16-bit WMMA fragment layout (ISA 7.12.2): lanes 0-15 hold K={0..7,16..23},
// lanes 16-31 hold K={8..15,24..31}. With the operand stored K-major per
// lane-row, each fragment is two contiguous 16-byte runs -> b128 loads.
// row_dw: dword pointer to this lane's row; base_dw: K-window start / 2.
// ---------------------------------------------------------------------------
__device__ __forceinline__ v16bf load_frag_kmajor(const unsigned* __restrict__ row_dw,
                                                  int base_dw, int lane) {
    union { v16bf bf; unsigned u[8]; } f;
    const int b0 = base_dw + ((lane & 16) ? 4 : 0);
    #pragma unroll
    for (int hp = 0; hp < 8; ++hp)
        f.u[hp] = row_dw[b0 + (hp & 3) + ((hp >> 2) << 3)];
    return f.bf;
}

// ---------------------------------------------------------------------------
// Kernel 1: LayerNorm over channels. x:[B,C,N] -> xn:[B*N,C] (f32 + bf16).
// ---------------------------------------------------------------------------
__global__ void __launch_bounds__(256) assa_ln_kernel(
    const float* __restrict__ x, const float* __restrict__ gamma,
    const float* __restrict__ beta, float* __restrict__ xn_f32,
    __bf16* __restrict__ xn_bf) {
    const int t = blockIdx.x * blockDim.x + threadIdx.x;   // token in [0, B*N)
    const int b = t >> 12;
    const int n = t & (NN - 1);
    const float* xb = x + ((size_t)b * CC) * NN + n;
    float s = 0.f, ss = 0.f;
    for (int c = 0; c < CC; ++c) {
        float v = xb[(size_t)c * NN];
        s += v; ss += v * v;
    }
    const float mu   = s * (1.0f / CC);
    const float var  = ss * (1.0f / CC) - mu * mu;
    const float rstd = rsqrtf(var + EPS);
    float*  of = xn_f32 + (size_t)t * CC;
    __bf16* ob = xn_bf  + (size_t)t * CC;
    for (int c = 0; c < CC; ++c) {
        float v = (xb[(size_t)c * NN] - mu) * rstd * gamma[c] + beta[c];
        of[c] = v;
        ob[c] = (__bf16)v;
    }
}

// ---------------------------------------------------------------------------
// Kernel 2: Wq/Wk/Wv fp32 -> bf16, stored TRANSPOSED (K-major): wbT[mat][col][k].
// ---------------------------------------------------------------------------
__global__ void __launch_bounds__(256) assa_cvtw_kernel(
    const float* __restrict__ wq, const float* __restrict__ wk,
    const float* __restrict__ wv, __bf16* __restrict__ wbT) {
    const int i   = blockIdx.x * blockDim.x + threadIdx.x; // [0, C*C)
    const int col = i >> 8, k = i & 255;
    wbT[i]               = (__bf16)wq[(size_t)k * CC + col];
    wbT[CC * CC + i]     = (__bf16)wk[(size_t)k * CC + col];
    wbT[2 * CC * CC + i] = (__bf16)wv[(size_t)k * CC + col];
}

// ---------------------------------------------------------------------------
// Kernel 3: QKV projection GEMM: [B*N,C] x [C,C] -> bf16.
// grid.x = B*N/16 row tiles, grid.y = matrix (0:Q 1:K 2:V).
// Q,K stored [token][c] (K-major for the S=QK^T contraction);
// V stored TRANSPOSED Vt[c][token] (K-major for the P*V contraction).
// ---------------------------------------------------------------------------
__global__ void __launch_bounds__(256) assa_qkv_kernel(
    const __bf16* __restrict__ xn_bf, const __bf16* __restrict__ wbT,
    __bf16* __restrict__ qkv) {
    __shared__ __bf16 As[16 * CC];                          // 8 KB A tile
    const int tid  = threadIdx.x;
    const int lane = tid & 31;
    const int wave = tid >> 5;
    const int row0 = blockIdx.x * 16;
    const int mat  = blockIdx.y;
    const __bf16* W   = wbT + (size_t)mat * CC * CC;        // [col][k]
    __bf16*       out = qkv + (size_t)mat * BN * CC;

    for (int i = tid; i < 16 * CC; i += 256)
        As[i] = xn_bf[(size_t)(row0 + (i >> 8)) * CC + (i & 255)];
    __syncthreads();

    const int m    = lane & 15;
    const int col0 = wave * 32 + (lane & 15);
    const int col1 = col0 + 16;
    const unsigned* Arow = (const unsigned*)(As + m * CC);
    const unsigned* W0   = (const unsigned*)(W + (size_t)col0 * CC);
    const unsigned* W1   = (const unsigned*)(W + (size_t)col1 * CC);

    v8f acc0 = {}, acc1 = {};
    #pragma unroll
    for (int kb = 0; kb < 8; ++kb) {                        // K = 256 in 8x32
        const v16bf a  = load_frag_kmajor(Arow, kb * 16, lane);
        const v16bf b0 = load_frag_kmajor(W0,   kb * 16, lane);
        const v16bf b1 = load_frag_kmajor(W1,   kb * 16, lane);
        acc0 = __builtin_amdgcn_wmma_f32_16x16x32_bf16(
            false, a, false, b0, (short)0, acc0, false, false);
        acc1 = __builtin_amdgcn_wmma_f32_16x16x32_bf16(
            false, a, false, b1, (short)0, acc1, false, false);
    }

    const int mbase = (lane >> 4) << 3;                     // D layout (7.12.2)
    if (mat != 2) {                                         // Q,K: [token][c]
        #pragma unroll
        for (int r = 0; r < 8; ++r) {
            out[(size_t)(row0 + mbase + r) * CC + col0] = (__bf16)acc0[r];
            out[(size_t)(row0 + mbase + r) * CC + col1] = (__bf16)acc1[r];
        }
    } else {                                                // V: Vt[c][token]
        #pragma unroll
        for (int r = 0; r < 8; ++r) {
            out[(size_t)col0 * BN + row0 + mbase + r] = (__bf16)acc0[r];
            out[(size_t)col1 * BN + row0 + mbase + r] = (__bf16)acc1[r];
        }
    }
}

// ---------------------------------------------------------------------------
// Kernel 4: fused attention. One block per (16 query rows, batch).
// Full score stripe S[16 x 4096] f32 in LDS (256 KB of the 320 KB WGP LDS):
// QK^T computed once, exact softmax stats, single-accumulator P*V pass with
// attn = a1*softmax(S) + a2*relu(S)^2. Epilogue transposes through LDS for
// contiguous [B,C,H,W] stores with the xn residual added.
// ---------------------------------------------------------------------------
__global__ void __launch_bounds__(256) assa_attn_kernel(
    const __bf16* __restrict__ Q, const __bf16* __restrict__ K,
    const __bf16* __restrict__ Vt, const float* __restrict__ xn_f32,
    const float* __restrict__ w1, const float* __restrict__ w2,
    float* __restrict__ out) {
    extern __shared__ char smem[];
    float*  S    = (float*)smem;                            // 16*4096 f32
    __bf16* Qs   = (__bf16*)(smem + 16 * NN * 4);           // 16*256 bf16
    float*  red  = (float*)(smem + 16 * NN * 4 + 16 * CC * 2); // 16*16
    float*  mrow = red + 256;
    float*  lrow = mrow + 16;
    float*  coef = lrow + 16;                               // a1, a2

    const int tid  = threadIdx.x;
    const int lane = tid & 31;
    const int wave = tid >> 5;
    const int b    = blockIdx.y;
    const int q0   = blockIdx.x * 16;
    const __bf16* Qb = Q + (size_t)b * NN * CC;
    const __bf16* Kb = K + (size_t)b * NN * CC;

    for (int i = tid; i < 16 * CC; i += 256)
        Qs[i] = Qb[(size_t)(q0 + (i >> 8)) * CC + (i & 255)];
    if (tid == 0) {
        const float e1 = __expf(w1[0]), e2 = __expf(w2[0]);
        coef[0] = e1 / (e1 + e2);
        coef[1] = e2 / (e1 + e2);
    }
    __syncthreads();

    // ---- Pass 1: S = Q K^T, keys split across the 8 waves ----
    const int m = lane & 15;
    const unsigned* Qrow = (const unsigned*)(Qs + m * CC);
    for (int kt = wave; kt < NN / 16; kt += 8) {
        const int key0 = kt * 16;
        const unsigned* Krow =
            (const unsigned*)(Kb + (size_t)(key0 + (lane & 15)) * CC);
        v8f acc = {};
        #pragma unroll
        for (int kb = 0; kb < 8; ++kb) {                    // contract C=256
            const v16bf a  = load_frag_kmajor(Qrow, kb * 16, lane);
            const v16bf bf = load_frag_kmajor(Krow, kb * 16, lane);
            acc = __builtin_amdgcn_wmma_f32_16x16x32_bf16(
                false, a, false, bf, (short)0, acc, false, false);
        }
        const int mbase = (lane >> 4) << 3;
        #pragma unroll
        for (int r = 0; r < 8; ++r)
            S[(mbase + r) * NN + key0 + (lane & 15)] = acc[r];
    }
    __syncthreads();

    // ---- Row stats: exact max and exp-sum over 4096 keys ----
    {
        const int r = tid >> 4, sidx = tid & 15;
        const float* Sr = S + r * NN + sidx * 256;
        float lm = -1e30f;
        for (int i = 0; i < 256; ++i) lm = fmaxf(lm, Sr[i]);
        red[r * 16 + sidx] = lm;
        __syncthreads();
        if (tid < 16) {
            float mm = -1e30f;
            for (int i = 0; i < 16; ++i) mm = fmaxf(mm, red[tid * 16 + i]);
            mrow[tid] = mm;
        }
        __syncthreads();
        const float rmv = mrow[r];
        float ls = 0.f;
        for (int i = 0; i < 256; ++i) ls += __expf(Sr[i] - rmv);
        red[r * 16 + sidx] = ls;
        __syncthreads();
        if (tid < 16) {
            float sm = 0.f;
            for (int i = 0; i < 16; ++i) sm += red[tid * 16 + i];
            lrow[tid] = sm;
        }
        __syncthreads();
    }

    // ---- Pass 2: out = (a1*softmax + a2*relu^2) @ V; wave owns 32 channels ----
    const float a1 = coef[0], a2 = coef[1];
    const int   cw = wave * 32;
    const float rmv = mrow[m], rl_inv = 1.0f / lrow[m];
    const float* Srow = S + m * NN;
    const __bf16* Vrow0 = Vt + (size_t)(cw + (lane & 15)) * BN + (size_t)b * NN;
    const unsigned* V0 = (const unsigned*)Vrow0;
    const unsigned* V1 = (const unsigned*)(Vrow0 + (size_t)16 * BN);

    v8f acc0 = {}, acc1 = {};
    for (int kt = 0; kt < NN / 32; ++kt) {
        union { v16bf bf; __bf16 h[16]; } p;
        #pragma unroll
        for (int hp = 0; hp < 8; ++hp) {                    // key pairs
            const int key = kt * 32 + ((lane & 16) ? 8 : 0) +
                            ((hp & 3) << 1) + ((hp >> 2) << 4);
            const float s0 = Srow[key], s1 = Srow[key + 1];
            const float r0 = fmaxf(s0, 0.f), r1 = fmaxf(s1, 0.f);
            p.h[2 * hp]     = (__bf16)(a1 * __expf(s0 - rmv) * rl_inv + a2 * r0 * r0);
            p.h[2 * hp + 1] = (__bf16)(a1 * __expf(s1 - rmv) * rl_inv + a2 * r1 * r1);
        }
        const v16bf v0 = load_frag_kmajor(V0, kt * 16, lane);
        const v16bf v1 = load_frag_kmajor(V1, kt * 16, lane);
        acc0 = __builtin_amdgcn_wmma_f32_16x16x32_bf16(
            false, p.bf, false, v0, (short)0, acc0, false, false);
        acc1 = __builtin_amdgcn_wmma_f32_16x16x32_bf16(
            false, p.bf, false, v1, (short)0, acc1, false, false);
    }

    // ---- Epilogue: stage (attn + residual) in LDS, contiguous stores ----
    __syncthreads();                                        // done reading S
    float* Sout = S;                                        // reuse: 16 x 256
    const float* xr = xn_f32 + ((size_t)b * NN + q0) * CC;
    const int mbase = (lane >> 4) << 3;
    #pragma unroll
    for (int r = 0; r < 8; ++r) {                           // coalesced xr reads
        const int rm2 = mbase + r;
        Sout[rm2 * CC + cw + (lane & 15)] =
            acc0[r] + xr[(size_t)rm2 * CC + cw + (lane & 15)];
        Sout[rm2 * CC + cw + 16 + (lane & 15)] =
            acc1[r] + xr[(size_t)rm2 * CC + cw + 16 + (lane & 15)];
    }
    __syncthreads();
    float* ob = out + (size_t)b * CC * NN + q0;
    for (int i = tid; i < 16 * CC; i += 256) {              // 64B segments
        const int c = i >> 4, nl = i & 15;
        ob[(size_t)c * NN + nl] = Sout[nl * CC + c];
    }
}

// ---------------------------------------------------------------------------
extern "C" void kernel_launch(void* const* d_in, const int* in_sizes, int n_in,
                              void* d_out, int out_size, void* d_ws, size_t ws_size,
                              hipStream_t stream) {
    const float* x     = (const float*)d_in[0];
    const float* gamma = (const float*)d_in[1];
    const float* beta  = (const float*)d_in[2];
    const float* Wq    = (const float*)d_in[3];
    const float* Wk    = (const float*)d_in[4];
    const float* Wv    = (const float*)d_in[5];
    const float* w1    = (const float*)d_in[6];
    const float* w2    = (const float*)d_in[7];
    float* out = (float*)d_out;

    // Workspace layout (bytes):
    //   xn_f32 : B*N*C*4   = 16,777,216
    //   xn_bf  : B*N*C*2   =  8,388,608
    //   wbT    : 3*C*C*2   =    393,216
    //   qkv    : 3*B*N*C*2 = 25,165,824   (slab2 = Vt[c][token])
    char* ws = (char*)d_ws;
    float*  xn_f32 = (float*)ws;
    __bf16* xn_bf  = (__bf16*)(ws + 16777216);
    __bf16* wbT    = (__bf16*)(ws + 16777216 + 8388608);
    __bf16* qkv    = (__bf16*)(ws + 16777216 + 8388608 + 393216);
    __bf16* Qm = qkv;
    __bf16* Km = qkv + (size_t)BN * CC;
    __bf16* Vt = qkv + 2 * (size_t)BN * CC;

    const size_t smem = (size_t)16 * NN * 4 + (size_t)16 * CC * 2 +
                        (256 + 16 + 16 + 8) * sizeof(float);
    hipFuncSetAttribute((const void*)assa_attn_kernel,
                        hipFuncAttributeMaxDynamicSharedMemorySize, (int)smem);

    assa_ln_kernel<<<dim3(BN / 256), dim3(256), 0, stream>>>(
        x, gamma, beta, xn_f32, xn_bf);
    assa_cvtw_kernel<<<dim3(CC * CC / 256), dim3(256), 0, stream>>>(
        Wq, Wk, Wv, wbT);
    assa_qkv_kernel<<<dim3(BN / 16, 3), dim3(256), 0, stream>>>(
        xn_bf, wbT, qkv);
    assa_attn_kernel<<<dim3(NN / 16, BB), dim3(256), smem, stream>>>(
        Qm, Km, Vt, xn_f32, w1, w2, out);
}
